// STFT_75118978007171
// MI455X (gfx1250) — matrix operand order repaired and checked by reading, not compiled, not run on hardware
//
#include <hip/hip_runtime.h>
#include <math.h>

typedef __attribute__((ext_vector_type(16))) _Float16 v16h;
typedef __attribute__((ext_vector_type(8)))  _Float16 v8h;
typedef __attribute__((ext_vector_type(8)))  float    v8f;

#define N_FFT   2048
#define HOP     512
#define FREQ    1025
#define PADF    1040            // 65 tiles * 16
#define NB      32
#define LSIG    220500
#define PAD     1024
#define XP_VALID (LSIG + 2*PAD) // 222548
#define XP_LEN   230912         // covers 448 frames (447*512+2048), multiple of 32
#define FRAMES  431
#define FTILES  65              // ceil(1025/16)
#define TPTILES 14              // ceil(431/32) -> 448 frames, 32 frames per wave

// ---- stage W (f32 -> f16, zero-pad rows 1025..1039) -------------------------
__global__ void prep_w(const float* __restrict__ src, _Float16* __restrict__ dst) {
    size_t total = (size_t)PADF * N_FFT;
    for (size_t i = (size_t)blockIdx.x * blockDim.x + threadIdx.x; i < total;
         i += (size_t)gridDim.x * blockDim.x) {
        size_t row = i / N_FFT;
        dst[i] = (row < FREQ) ? (_Float16)src[i] : (_Float16)0.0f;
    }
}

// ---- stage x: reflect-pad + f32 -> f16, zero tail ---------------------------
__global__ void prep_x(const float* __restrict__ x, _Float16* __restrict__ xp) {
    size_t total = (size_t)NB * XP_LEN;
    for (size_t i = (size_t)blockIdx.x * blockDim.x + threadIdx.x; i < total;
         i += (size_t)gridDim.x * blockDim.x) {
        int b = (int)(i / XP_LEN);
        int p = (int)(i % XP_LEN);
        float v = 0.0f;
        if (p < XP_VALID) {
            int j = p - PAD;
            if (j < 0) j = -j;                          // left reflect
            else if (j >= LSIG) j = 2 * (LSIG - 1) - j; // right reflect
            v = x[(size_t)b * LSIG + j];
        }
        xp[i] = (_Float16)v;
    }
}

// ---- main GEMM: one wave computes a 16(freq) x 32(frame) tile ---------------
// A (16x32 f16): lane m=lane&15, half=lane>>4 holds K = half*16+[0..15], contiguous in memory
// B (32x16 f16): lane n=lane&15, half       holds K = half*16+[0..15], contiguous in memory
// C/D (16x16 f32): vgpr r, lane -> (M = 8*half + r, N = lane&15)
// A_sin/A_cos registers amortized over two B columns: 8 b128 loads / 4 WMMAs per K-step.
__global__ void __launch_bounds__(128) stft_wmma(
    const _Float16* __restrict__ Ws, const _Float16* __restrict__ Wc,
    const _Float16* __restrict__ Xp, float* __restrict__ out)
{
    const int lane = threadIdx.x & 31;
    const int wave = threadIdx.x >> 5;
    const int half = lane >> 4;
    const int mn   = lane & 15;

    const int F0    = blockIdx.x * 16;          // 4 waves in a block share this A tile
    const int tTile = blockIdx.y * 4 + wave;
    if (tTile >= TPTILES) return;               // uniform per-wave exit, EXEC stays all-1s
    const int T0 = tTile * 32;
    const int b  = blockIdx.z;

    const _Float16* arow_s = Ws + (size_t)(F0 + mn) * N_FFT + half * 16;
    const _Float16* arow_c = Wc + (size_t)(F0 + mn) * N_FFT + half * 16;
    const _Float16* bcol   = Xp + (size_t)b * XP_LEN + (size_t)(T0 + mn) * HOP + half * 16;

    v8f acc_s0 = {};
    v8f acc_s1 = {};
    v8f acc_c0 = {};
    v8f acc_c1 = {};

#pragma unroll 2
    for (int kk = 0; kk < N_FFT; kk += 32) {
        v8h  as0 = *(const v8h*)(arow_s + kk);        // K = kk + half*16 + [0..7]
        v8h  as1 = *(const v8h*)(arow_s + kk + 8);    // K = kk + half*16 + 8 + [0..7]
        v8h  ac0 = *(const v8h*)(arow_c + kk);
        v8h  ac1 = *(const v8h*)(arow_c + kk + 8);
        v16h b0  = *(const v16h*)(bcol + kk);              // frames T0+[0..15]
        v16h b1  = *(const v16h*)(bcol + kk + 16 * HOP);   // frames T0+16+[0..15]
        v16h a_s = __builtin_shufflevector(as0, as1, 0,1,2,3,4,5,6,7,8,9,10,11,12,13,14,15);
        v16h a_c = __builtin_shufflevector(ac0, ac1, 0,1,2,3,4,5,6,7,8,9,10,11,12,13,14,15);
        acc_s0 = __builtin_amdgcn_wmma_f32_16x16x32_f16(
            false, a_s, false, b0, (short)0, acc_s0, false, false);
        acc_s1 = __builtin_amdgcn_wmma_f32_16x16x32_f16(
            false, a_s, false, b1, (short)0, acc_s1, false, false);
        acc_c0 = __builtin_amdgcn_wmma_f32_16x16x32_f16(
            false, a_c, false, b0, (short)0, acc_c0, false, false);
        acc_c1 = __builtin_amdgcn_wmma_f32_16x16x32_f16(
            false, a_c, false, b1, (short)0, acc_c1, false, false);
    }

    // magnitude + store (guards only here, after all WMMAs)
    float* obb = out + (size_t)b * FREQ * FRAMES;
    {
        const int t = T0 + mn;                     // column set 0
        if (t < FRAMES) {
#pragma unroll
            for (int r = 0; r < 8; ++r) {
                const int f = F0 + half * 8 + r;
                if (f < FREQ) {
                    const float s = acc_s0[r];
                    const float c = acc_c0[r];
                    obb[(size_t)f * FRAMES + t] = sqrtf(s * s + c * c);
                }
            }
        }
    }
    {
        const int t = T0 + 16 + mn;                // column set 1
        if (t < FRAMES) {
#pragma unroll
            for (int r = 0; r < 8; ++r) {
                const int f = F0 + half * 8 + r;
                if (f < FREQ) {
                    const float s = acc_s1[r];
                    const float c = acc_c1[r];
                    obb[(size_t)f * FRAMES + t] = sqrtf(s * s + c * c);
                }
            }
        }
    }
}

extern "C" void kernel_launch(void* const* d_in, const int* in_sizes, int n_in,
                              void* d_out, int out_size, void* d_ws, size_t ws_size,
                              hipStream_t stream) {
    (void)in_sizes; (void)n_in; (void)out_size; (void)ws_size;
    const float* x    = (const float*)d_in[0];   // (32, 220500) f32
    const float* wsin = (const float*)d_in[1];   // (1025, 1, 2048) f32
    const float* wcos = (const float*)d_in[2];   // (1025, 1, 2048) f32
    float* out = (float*)d_out;                  // (32, 1025, 431) f32

    // workspace layout (all offsets 32B-aligned):
    _Float16* ws16 = (_Float16*)d_ws;                         // 1040*2048 f16
    _Float16* wc16 = ws16 + (size_t)PADF * N_FFT;             // 1040*2048 f16
    _Float16* xp16 = wc16 + (size_t)PADF * N_FFT;             // 32*230912 f16
    // total ~23.3 MB

    {
        const int threads = 256;
        const int blocksW = (int)(((size_t)PADF * N_FFT + threads - 1) / threads);
        prep_w<<<blocksW, threads, 0, stream>>>(wsin, ws16);
        prep_w<<<blocksW, threads, 0, stream>>>(wcos, wc16);
        const int blocksX = (int)(((size_t)NB * XP_LEN + threads - 1) / threads);
        prep_x<<<blocksX, threads, 0, stream>>>(x, xp16);
    }

    dim3 grid(FTILES, (TPTILES + 3) / 4, NB);   // 65 x 4 x 32 blocks, 4 waves each
    stft_wmma<<<grid, 128, 0, stream>>>(ws16, wc16, xp16, out);
}
